// GAT_50818053046713
// MI455X (gfx1250) — matrix operand order repaired and checked by reading, not compiled, not run on hardware
//
#include <hip/hip_runtime.h>
#include <hip/hip_bf16.h>
#include <math.h>

typedef __attribute__((ext_vector_type(16))) _Float16 v16h;
typedef __attribute__((ext_vector_type(8)))  _Float16 v8h;
typedef __attribute__((ext_vector_type(8)))  float    v8f;

#define NEG_SLOPE 0.2f

// ---------------- precision / layout conversion ----------------
__global__ void k_f32_to_f16(const float* __restrict__ in, _Float16* __restrict__ out, int n) {
  int t = blockIdx.x * blockDim.x + threadIdx.x;
  if (t < n) out[t] = (_Float16)in[t];
}

// W[k*N + n] (row-major [K,N]) -> Wt[n*K + k] (row-major [N,K]) in f16
__global__ void k_transpose_f16(const float* __restrict__ W, _Float16* __restrict__ Wt, int K, int N) {
  int t = blockIdx.x * blockDim.x + threadIdx.x;
  if (t >= K * N) return;
  int k = t / N, n = t - k * N;
  Wt[n * K + k] = (_Float16)W[t];
}

__global__ void k_fill(float* __restrict__ p, float v, int n) {
  int t = blockIdx.x * blockDim.x + threadIdx.x;
  if (t < n) p[t] = v;
}

// ---------------- WMMA GEMM: C[M,128] = A[M,128] * B[128,128] ----------------
// A: f16 row-major [M,128]; Bt: f16 row-major [N=128, K=128] (pre-transposed weights)
// block = 256 threads = 8 waves. The block's 16x128 A stripe is staged once in
// LDS (4KB) and shared by all 8 waves; wave w computes the 16x16 tile at cols
// w*16. K=128 -> 4 x v_wmma_f32_16x16x32_f16 with f32 accumulation.
__global__ __launch_bounds__(256)
void k_wmma_gemm_hc128(const _Float16* __restrict__ A,
                       const _Float16* __restrict__ Bt,
                       float* __restrict__ C, int M) {
  __shared__ __align__(16) _Float16 As[16 * 128];

  const int t  = threadIdx.x;
  const int m0 = blockIdx.x * 16;

  // cooperative copy: 16 rows x 256B; each thread moves one 16B chunk
  {
    int row   = t >> 4;          // 0..15
    int chunk = t & 15;          // 0..15 (16B each)
    int grow  = m0 + row;
    if (grow >= M) grow = M - 1; // clamp for ragged tail (stores are guarded)
    const float4* src = (const float4*)(A + (size_t)grow * 128 + chunk * 8);
    *(float4*)(As + row * 128 + chunk * 8) = *src;
  }
  __syncthreads();

  const int lane  = t & 31;
  const int n0    = (t >> 5) * 16;   // column tile of this wave
  const int khalf = lane >> 4;       // which K half-group this lane holds
  const int l15   = lane & 15;

  v8f acc = {};
#pragma unroll
  for (int kb = 0; kb < 4; ++kb) {
    const int k0 = kb * 32;
    // A 16x32 f16 fragment (contiguous k order, same as B): lane m=l15, khalf selects K subgroups.
    // halves 0..7  <- K = k0 + 16*khalf + 0..7
    // halves 8..15 <- K = k0 + 16*khalf + 8 + 0..7
    const _Float16* pa = As + l15 * 128 + k0 + 16 * khalf;
    v8h a0 = *(const v8h*)pa;
    v8h a1 = *(const v8h*)(pa + 8);
    v16h av;
#pragma unroll
    for (int i = 0; i < 8; ++i) { av[i] = a0[i]; av[8 + i] = a1[i]; }
    // B 32x16 f16 fragment: lane n=n0+l15, halves e <- K = k0 + 16*khalf + e
    v16h bv = *(const v16h*)(Bt + (size_t)(n0 + l15) * 128 + k0 + 16 * khalf);
    acc = __builtin_amdgcn_wmma_f32_16x16x32_f16(false, av, false, bv,
                                                 (short)0, acc, false, false);
  }

  // C/D layout: VGPR r -> row m0 + r + 8*khalf, col n0 + l15
  if (m0 + 16 <= M) {               // fast path: whole tile in-bounds, no guards
    float* cp = C + (size_t)(m0 + 8 * khalf) * 128 + n0 + l15;
#pragma unroll
    for (int r = 0; r < 8; ++r) cp[(size_t)r * 128] = acc[r];
  } else {
#pragma unroll
    for (int r = 0; r < 8; ++r) {
      int row = m0 + r + 8 * khalf;
      if (row < M) C[(size_t)row * 128 + n0 + l15] = acc[r];
    }
  }
}

// ---------------- attention logits per node ----------------
// xh: [N, H*C]; att*: [H, C]; out aS/aD: [N*H]
__global__ void k_compute_a(const float* __restrict__ xh,
                            const float* __restrict__ attS, const float* __restrict__ attD,
                            float* __restrict__ aS, float* __restrict__ aD,
                            int N, int H, int C) {
  int t = blockIdx.x * blockDim.x + threadIdx.x;
  if (t >= N * H) return;
  int h = t % H;
  const float* row = xh + (size_t)t * C;        // (n*H+h)*C == n*(H*C) + h*C
  float s = 0.f, d = 0.f;
  for (int c = 0; c < C; ++c) {
    float v = row[c];
    s += v * attS[h * C + c];
    d += v * attD[h * C + c];
  }
  aS[t] = s; aD[t] = d;
}

// ---------------- edge passes ----------------
__device__ __forceinline__ float lrelu(float a) { return a > 0.f ? a : NEG_SLOPE * a; }

// CAS-free float atomic max (location must be initialized to -inf)
__device__ __forceinline__ void atomicMaxF32(float* addr, float v) {
  if (v >= 0.f) atomicMax((int*)addr, __float_as_int(v));
  else          atomicMin((unsigned int*)addr, __float_as_uint(v));
}

__global__ void k_edge_max(const int* __restrict__ src, const int* __restrict__ dst,
                           const float* __restrict__ aS, const float* __restrict__ aD,
                           float* __restrict__ mmax, int E, int N, int H) {
  int t = blockIdx.x * blockDim.x + threadIdx.x;
  int tot = (E + N) * H;
  if (t >= tot) return;
  int e = t / H, h = t - e * H;
  int s = e < E ? src[e] : (e - E);
  int d = e < E ? dst[e] : (e - E);
  float a = lrelu(aS[s * H + h] + aD[d * H + h]);
  atomicMaxF32(&mmax[d * H + h], a);
}

__global__ void k_edge_exp(const int* __restrict__ src, const int* __restrict__ dst,
                           const float* __restrict__ aS, const float* __restrict__ aD,
                           const float* __restrict__ mmax,
                           float* __restrict__ ecoef, float* __restrict__ ssum,
                           int E, int N, int H) {
  int t = blockIdx.x * blockDim.x + threadIdx.x;
  int tot = (E + N) * H;
  if (t >= tot) return;
  int e = t / H, h = t - e * H;
  int s = e < E ? src[e] : (e - E);
  int d = e < E ? dst[e] : (e - E);
  float a  = lrelu(aS[s * H + h] + aD[d * H + h]);
  float ev = expf(a - mmax[d * H + h]);
  ecoef[t] = ev;
  atomicAdd(&ssum[d * H + h], ev);
}

// wave-per-edge scatter: accum[dst, 0:128] += ecoef[e,h] * xh[src, 0:128]
// lane l handles components c = 4l..4l+3 (all inside one head since C % 4 == 0)
__global__ void k_scatter(const float* __restrict__ xh, const float* __restrict__ ecoef,
                          const int* __restrict__ src, const int* __restrict__ dst,
                          float* __restrict__ accum, int E, int N, int H, int C) {
  int gw   = (blockIdx.x * blockDim.x + threadIdx.x) >> 5;  // global wave id == edge
  int lane = threadIdx.x & 31;
  int Etot = E + N;
  if (gw >= Etot) return;
  int s = gw < E ? src[gw] : (gw - E);
  int d = gw < E ? dst[gw] : (gw - E);
  int c = lane * 4;
  int h = c / C;
  float coef = ecoef[(size_t)gw * H + h];
  float4 v = *(const float4*)(xh + (size_t)s * 128 + c);
  float* o = accum + (size_t)d * 128 + c;
  atomicAdd(o + 0, v.x * coef);
  atomicAdd(o + 1, v.y * coef);
  atomicAdd(o + 2, v.z * coef);
  atomicAdd(o + 3, v.w * coef);
}

// normalize by segment sum, add bias, optional ELU; emit f16 (next layer) or f32 (d_out)
__global__ void k_finalize(const float* __restrict__ accum, const float* __restrict__ ssum,
                           const float* __restrict__ bias,
                           _Float16* __restrict__ of16, float* __restrict__ of32,
                           int N, int H, int C, int do_elu) {
  int t = blockIdx.x * blockDim.x + threadIdx.x;
  if (t >= N * 128) return;
  int n = t >> 7, c = t & 127;
  int h = c / C;
  float v = accum[t] / ssum[n * H + h] + bias[c];
  if (do_elu) v = v > 0.f ? v : expm1f(v);
  if (of16) of16[t] = (_Float16)v;
  else      of32[t] = v;
}

// ---------------- launcher ----------------
extern "C" void kernel_launch(void* const* d_in, const int* in_sizes, int n_in,
                              void* d_out, int out_size, void* d_ws, size_t ws_size,
                              hipStream_t stream) {
  (void)n_in; (void)out_size; (void)ws_size;
  const float* x   = (const float*)d_in[0];
  const int*   src = (const int*)d_in[1];
  const int*   dst = (const int*)d_in[2];
  const float* W1  = (const float*)d_in[3];
  const float* as1 = (const float*)d_in[4];
  const float* ad1 = (const float*)d_in[5];
  const float* b1  = (const float*)d_in[6];
  const float* W2  = (const float*)d_in[7];
  const float* as2 = (const float*)d_in[8];
  const float* ad2 = (const float*)d_in[9];
  const float* b2  = (const float*)d_in[10];

  const int N    = in_sizes[0] / 128;
  const int E    = in_sizes[1];
  const int Etot = E + N;   // self-loops appended

  // workspace carve (256B aligned); buffers are reused across the two layers
  char* w = (char*)d_ws;
  auto carve = [&](size_t bytes) { char* p = w; w += (bytes + 255) & ~(size_t)255; return p; };
  _Float16* xf16  = (_Float16*)carve((size_t)N * 128 * 2);  // layer input f16 (reused: h_f16)
  _Float16* W1t   = (_Float16*)carve(128 * 128 * 2);
  _Float16* W2t   = (_Float16*)carve(128 * 128 * 2);
  float*    xh    = (float*)carve((size_t)N * 128 * 4);
  float*    aS    = (float*)carve((size_t)N * 8 * 4);
  float*    aD    = (float*)carve((size_t)N * 8 * 4);
  float*    mmax  = (float*)carve((size_t)N * 8 * 4);
  float*    ssum  = (float*)carve((size_t)N * 8 * 4);
  float*    ecoef = (float*)carve((size_t)Etot * 8 * 4);
  float*    accum = (float*)carve((size_t)N * 128 * 4);

  const int B = 256;
  auto G = [](long n, int b) { return (unsigned)((n + b - 1) / b); };

  k_f32_to_f16<<<G((long)N * 128, B), B, 0, stream>>>(x, xf16, N * 128);
  k_transpose_f16<<<G(128 * 128, B), B, 0, stream>>>(W1, W1t, 128, 128);
  k_transpose_f16<<<G(128 * 128, B), B, 0, stream>>>(W2, W2t, 128, 128);

  for (int layer = 0; layer < 2; ++layer) {
    const int H = layer ? 1 : 8;
    const int C = layer ? 128 : 16;
    const _Float16* Bt   = layer ? W2t : W1t;
    const float*    attS = layer ? as2 : as1;
    const float*    attD = layer ? ad2 : ad1;
    const float*    bias = layer ? b2  : b1;

    k_wmma_gemm_hc128<<<(N + 15) / 16, 256, 0, stream>>>(xf16, Bt, xh, N);
    k_compute_a<<<G((long)N * H, B), B, 0, stream>>>(xh, attS, attD, aS, aD, N, H, C);
    k_fill<<<G((long)N * H, B), B, 0, stream>>>(mmax, -INFINITY, N * H);
    k_fill<<<G((long)N * H, B), B, 0, stream>>>(ssum, 0.f, N * H);
    k_fill<<<G((long)N * 128, B), B, 0, stream>>>(accum, 0.f, N * 128);
    k_edge_max<<<G((long)Etot * H, B), B, 0, stream>>>(src, dst, aS, aD, mmax, E, N, H);
    k_edge_exp<<<G((long)Etot * H, B), B, 0, stream>>>(src, dst, aS, aD, mmax, ecoef, ssum, E, N, H);
    k_scatter<<<G((long)Etot * 32, B), B, 0, stream>>>(xh, ecoef, src, dst, accum, E, N, H, C);
    k_finalize<<<G((long)N * 128, B), B, 0, stream>>>(
        accum, ssum, bias,
        layer ? (_Float16*)nullptr : xf16,
        layer ? (float*)d_out : (float*)nullptr,
        N, H, C, layer == 0 ? 1 : 0);
  }
}
